// ModelBase_37924561223724
// MI455X (gfx1250) — matrix operand order repaired and checked by reading, not compiled, not run on hardware
//
#include <hip/hip_runtime.h>
#include <hip/hip_bf16.h>
#include <stdint.h>

// ---- sizes (fixed by the reference) ----
#define B 1024
#define L 256
#define H 128
#define NUM_TEAMS 1000
#define NTPAD 1024          // teams padded to multiple of 16 for WMMA tiles
#define OUT_IN 517          // 4H + 2 + 3

typedef __attribute__((ext_vector_type(16))) __bf16    v16bf;
typedef __attribute__((ext_vector_type(8)))  float     v8f;
typedef __attribute__((ext_vector_type(4)))  uint32_t  u32x4;

union FragBF {
    v16bf v;
    u32x4 q[2];   // two 16-byte halves; q[k].i = VGPR(4k+i) = packed bf16 pair
};

// ---------------------------------------------------------------------------
// Kernel 1: team_embed f32 (1000x128) -> bf16 (1024x128, zero padded), RNE.
// ---------------------------------------------------------------------------
__global__ __launch_bounds__(256) void prep_bf16_kernel(
        const float* __restrict__ team, uint16_t* __restrict__ Tbf) {
    int i = blockIdx.x * 256 + threadIdx.x;          // 0 .. 1024*128-1
    if (i >= NTPAD * H) return;
    int row = i >> 7;
    float f = (row < NUM_TEAMS) ? team[i] : 0.0f;    // row-major, stride 128
    uint32_t u = __float_as_uint(f);
    Tbf[i] = (uint16_t)((u + 0x7FFFu + ((u >> 16) & 1u)) >> 16);  // RNE
}

// ---------------------------------------------------------------------------
// Kernel 2: TG = T_bf16 @ T_bf16^T  (1024x1024, f32 accum) via WMMA.
// One wave (32 lanes) per 16x16 tile; K=128 -> 4x wmma_f32_16x16x32_bf16.
// Per CDNA5 ISA 7.12.2 each lane's fragment bytes are CONTIGUOUS in the
// row-major bf16 table, so every fragment is exactly 2x global_load_b128:
//   A (16x32), lane half h: K = kc + 8h + [0..7]  and +16       (16B each)
//   B (32x16) = T^T, lane half h: K = kc + 8h + [0..7]  and +16 (2x16B)
// All 16 b128 loads for K=128 are issued up front, then 4 WMMAs back-to-back.
// ---------------------------------------------------------------------------
__global__ __launch_bounds__(32) void gram_wmma_kernel(
        const uint16_t* __restrict__ Tbf, float* __restrict__ TG) {
    const int m0   = blockIdx.x * 16;
    const int n0   = blockIdx.y * 16;
    const int lane = threadIdx.x;
    const int loc  = lane & 15;        // M for A-frag, N for B-frag
    const int h    = lane >> 4;        // lane half selector

    // per-lane base pointers (dword units; bf16 pairs)
    const uint32_t* T32  = (const uint32_t*)Tbf;
    const uint32_t* Arow = T32 + ((((m0 + loc) * H) + (h << 3)) >> 1);
    const uint32_t* Brow = T32 + ((((n0 + loc) * H) + (h << 3)) >> 1);

    FragBF a[4], b[4];
    #pragma unroll
    for (int t = 0; t < 4; ++t) {                  // kc = 32*t
        const int kd = (t << 5) >> 1;              // kc/2 dwords
        a[t].q[0] = *(const u32x4*)(Arow + kd);        // K = kc+8h   .. +7
        a[t].q[1] = *(const u32x4*)(Arow + kd + 8);    // K = kc+8h+16.. +23
        b[t].q[0] = *(const u32x4*)(Brow + kd);        // K = kc+8h   .. +7
        b[t].q[1] = *(const u32x4*)(Brow + kd + 8);    // K = kc+8h+16.. +23
    }

    v8f c = {};
    #pragma unroll
    for (int t = 0; t < 4; ++t)
        c = __builtin_amdgcn_wmma_f32_16x16x32_bf16(
                /*neg_a=*/false, a[t].v, /*neg_b=*/false, b[t].v,
                /*c_mod=*/(short)0, c, /*reuse_a=*/false, /*reuse_b=*/false);

    // D layout: VGPR r holds M = r + 8*h ; N = lane & 15
    #pragma unroll
    for (int r = 0; r < 8; ++r)
        TG[(size_t)(m0 + r + (h << 3)) * NTPAD + (n0 + loc)] = c[r];
}

// ---------------------------------------------------------------------------
// Kernel 3: per-batch attention + head. One 256-thread block per b.
// scores via Gram-table gathers, LDS softmax, 514-dim weighted embedding sum,
// 517x3 output head with warp reductions.
// ---------------------------------------------------------------------------
__global__ __launch_bounds__(256) void attention_kernel(
        const float* __restrict__ TG,
        const float* __restrict__ team_embed,
        const float* __restrict__ venue_embed,
        const float* __restrict__ result_embed,
        const float* __restrict__ W_out,
        const float* __restrict__ b_out,
        const float* __restrict__ goals_for,
        const float* __restrict__ goals_against,
        const float* __restrict__ stats,
        const int* __restrict__ venue,
        const int* __restrict__ team,
        const int* __restrict__ opponent,
        const int* __restrict__ result,
        const int* __restrict__ next_venue,
        const int* __restrict__ next_team,
        const int* __restrict__ next_opponent,
        float* __restrict__ out) {
    const int b    = blockIdx.x;
    const int t    = threadIdx.x;
    const int lane = t & 31;
    const int wid  = t >> 5;

    __shared__ int   s_ven[L], s_team[L], s_opp[L], s_res[L];
    __shared__ float s_gf[L], s_ga[L];
    __shared__ float s_attn[L];
    __shared__ float s_repr[OUT_IN + 3];
    __shared__ float s_vg[3];
    __shared__ float red[8];

    // stage indices + goal scalars
    s_ven[t]  = venue[b * L + t];
    s_team[t] = team[b * L + t];
    s_opp[t]  = opponent[b * L + t];
    s_res[t]  = result[b * L + t];
    s_gf[t]   = goals_for[b * L + t];
    s_ga[t]   = goals_against[b * L + t];

    const int nv = next_venue[b];
    const int nt = next_team[b];
    const int no = next_opponent[b];

    // venue "Gram" column (3 dots of length 128, trivial)
    if (t < 3) {
        float acc = 0.0f;
        #pragma unroll 4
        for (int k = 0; k < H; ++k)
            acc += venue_embed[t * H + k] * venue_embed[nv * H + k];
        s_vg[t] = acc;
    }
    __syncthreads();

    // scores from L2-resident Gram tables
    const float scale = 0.051031036307982884f;  // 1/sqrt(384)
    float sc = (s_vg[s_ven[t]]
              + TG[(size_t)s_team[t] * NTPAD + nt]
              + TG[(size_t)s_opp[t]  * NTPAD + no]) * scale;

    // softmax over L=256 (8 waves)
    float m = sc;
    #pragma unroll
    for (int off = 16; off; off >>= 1) m = fmaxf(m, __shfl_xor(m, off));
    if (lane == 0) red[wid] = m;
    __syncthreads();
    float gmax = red[0];
    #pragma unroll
    for (int i = 1; i < 8; ++i) gmax = fmaxf(gmax, red[i]);
    __syncthreads();

    float e = __expf(sc - gmax);
    float s = e;
    #pragma unroll
    for (int off = 16; off; off >>= 1) s += __shfl_xor(s, off);
    if (lane == 0) red[wid] = s;
    __syncthreads();
    float gsum = red[0] + red[1] + red[2] + red[3]
               + red[4] + red[5] + red[6] + red[7];
    s_attn[t] = e / gsum;
    __syncthreads();

    // past_repr[0:512]: attention-weighted sums of gathered embedding rows.
    // threads (t%128)=col, (t/128) picks chunk parity -> each thread does 2 chunks.
    {
        const int col  = t & 127;
        const int half = t >> 7;
        for (int c = half; c < 4; c += 2) {
            const float* tab;
            const int*   idx;
            if      (c == 0) { tab = venue_embed;  idx = s_ven;  }
            else if (c == 1) { tab = team_embed;   idx = s_team; }
            else if (c == 2) { tab = team_embed;   idx = s_opp;  }
            else             { tab = result_embed; idx = s_res;  }
            float acc = 0.0f;
            #pragma unroll 4
            for (int l = 0; l < L; ++l)
                acc += s_attn[l] * tab[idx[l] * H + col];
            s_repr[c * H + col] = acc;
        }
    }
    // goal scalars and stats tail
    if (t == 0) {
        float acc = 0.0f;
        #pragma unroll 4
        for (int l = 0; l < L; ++l) acc += s_attn[l] * s_gf[l];
        s_repr[4 * H] = acc;
    } else if (t == 1) {
        float acc = 0.0f;
        #pragma unroll 4
        for (int l = 0; l < L; ++l) acc += s_attn[l] * s_ga[l];
        s_repr[4 * H + 1] = acc;
    } else if (t >= 2 && t < 5) {
        s_repr[4 * H + t] = stats[b * 3 + (t - 2)];  // 514..516
    }
    __syncthreads();

    // logits = repr(517) @ W_out(517,3) + b_out : one warp per output column
    if (wid < 3) {
        const int r = wid;
        float acc = 0.0f;
        for (int d = lane; d < OUT_IN; d += 32)
            acc += s_repr[d] * W_out[d * 3 + r];
        #pragma unroll
        for (int off = 16; off; off >>= 1) acc += __shfl_xor(acc, off);
        if (lane == 0) out[b * 3 + r] = acc + b_out[r];
    }
}

// ---------------------------------------------------------------------------
extern "C" void kernel_launch(void* const* d_in, const int* in_sizes, int n_in,
                              void* d_out, int out_size, void* d_ws, size_t ws_size,
                              hipStream_t stream) {
    const float* team_embed    = (const float*)d_in[0];
    const float* venue_embed   = (const float*)d_in[1];
    const float* result_embed  = (const float*)d_in[2];
    const float* W_out         = (const float*)d_in[3];
    const float* b_out         = (const float*)d_in[4];
    const float* goals_for     = (const float*)d_in[5];
    const float* goals_against = (const float*)d_in[6];
    const float* stats         = (const float*)d_in[7];
    const int*   venue         = (const int*)d_in[8];
    const int*   team          = (const int*)d_in[9];
    const int*   opponent      = (const int*)d_in[10];
    const int*   result        = (const int*)d_in[11];
    const int*   next_venue    = (const int*)d_in[12];
    const int*   next_team     = (const int*)d_in[13];
    const int*   next_opponent = (const int*)d_in[14];
    float*       out           = (float*)d_out;

    // workspace: [ TG: 1024*1024 f32 = 4MB | Tbf: 1024*128 bf16 = 256KB ]
    float*    TG  = (float*)d_ws;
    uint16_t* Tbf = (uint16_t*)((char*)d_ws + (size_t)NTPAD * NTPAD * sizeof(float));

    prep_bf16_kernel<<<(NTPAD * H + 255) / 256, 256, 0, stream>>>(team_embed, Tbf);

    dim3 gg(NTPAD / 16, NTPAD / 16);
    gram_wmma_kernel<<<gg, 32, 0, stream>>>(Tbf, TG);

    attention_kernel<<<B, 256, 0, stream>>>(
        TG, team_embed, venue_embed, result_embed, W_out, b_out,
        goals_for, goals_against, stats,
        venue, team, opponent, result,
        next_venue, next_team, next_opponent, out);
}
